// GPT2CustomModel_8048768713030
// MI455X (gfx1250) — matrix operand order repaired and checked by reading, not compiled, not run on hardware
//
#include <hip/hip_runtime.h>
#include <hip/hip_bf16.h>

// ---------------- types / constants ----------------
typedef __attribute__((ext_vector_type(16))) __bf16       v16bf;
typedef __attribute__((ext_vector_type(8)))  float        v8f;
typedef __attribute__((ext_vector_type(4)))  unsigned     uv4;
typedef __attribute__((ext_vector_type(4)))  float        fv4;
typedef __attribute__((ext_vector_type(4)))  unsigned int tdm_v4u;
typedef __attribute__((ext_vector_type(8)))  int          tdm_v8i;
typedef __attribute__((ext_vector_type(4)))  int          tdm_v4i;

#define DMODEL 768
#define SEQ    1024
#define NROWS  2048      // B * S
#define NHEAD  12
#define HDIM   64
#define LDQKV  2304      // 3*DMODEL
#define NLAYER 12
#define FFDIM  3072
#define VOCAB  50264
#define KW     128       // GEMM K-chunk staged in LDS per TDM transfer

#if __has_builtin(__builtin_amdgcn_tensor_load_to_lds)
#define HAVE_TDM 1
#else
#define HAVE_TDM 0
#endif

// 16x16 bf16 WMMA fragment (8 VGPRs worth per lane)
union BFrag {
    v16bf          v;
    uv4            q[2];
    unsigned       u[8];
    unsigned short s[16];
};

__device__ inline v8f wmma_bf16(const BFrag& a, const BFrag& b, v8f c) {
    return __builtin_amdgcn_wmma_f32_16x16x32_bf16(false, a.v, false, b.v,
                                                   (short)0, c, false, false);
}

__device__ inline unsigned short to_bf16_bits(float a) {
    return __builtin_bit_cast(unsigned short, (__bf16)a);
}
__device__ inline unsigned pack2_bf16(float a, float b) {
    unsigned short lo = to_bf16_bits(a), hi = to_bf16_bits(b);
    return (unsigned)lo | ((unsigned)hi << 16);
}
__device__ inline v8f vzero8() {
    v8f v;
#pragma unroll
    for (int i = 0; i < 8; ++i) v[i] = 0.0f;
    return v;
}

#if HAVE_TDM
// TDM: DMA a 64-row x KW-col bf16 tile (row stride = ldk elements) from global to LDS.
// D# built per CDNA5 ISA 8.3/8.4: group0 {count, lds_addr, global_addr, type=2},
// group1 {data_size=1(2B), tensor dims/strides, tile dims}, groups 2/3 zero (2-D tensor).
// This toolchain exposes the 6-arg builtin: (v4u, v8i, v4i, v4i, v8i, cpol).
__device__ inline void tdm_load_tile(const unsigned short* gsrc, unsigned short* ldst,
                                     int ldk) {
    unsigned long long ga = (unsigned long long)(uintptr_t)gsrc;   // global byte address
    unsigned lds = (unsigned)(uintptr_t)ldst;                      // LDS byte offset (addr[31:0])
    tdm_v4u g0;
    g0[0] = 1u;                                             // count=1, no gather
    g0[1] = lds;                                            // lds_addr
    g0[2] = (unsigned)(ga & 0xFFFFFFFFu);                   // global_addr[31:0]
    g0[3] = (unsigned)((ga >> 32) & 0x01FFFFFFu)            // global_addr[56:32]
          | (2u << 30);                                     // type=2 ("image")
    unsigned long long td0 = (unsigned long long)ldk;       // tensor_dim0 (elements)
    unsigned long long td1 = (unsigned long long)NROWS;     // tensor_dim1 (rows)
    unsigned long long st0 = (unsigned long long)ldk;       // tensor_dim0_stride (elements)
    tdm_v8i g1;
    g1[0] = (int)(1u << 16);                                // data_size=1 -> 2 bytes
    g1[1] = (int)((unsigned)(td0 & 0xFFFFu) << 16);         // atomic_barrier_addr=0 | td0[15:0]
    g1[2] = (int)(((td0 >> 16) & 0xFFFFu) | ((td1 & 0xFFFFu) << 16));
    g1[3] = (int)(((td1 >> 16) & 0xFFFFu) | ((unsigned)KW << 16)); // tile_dim0 = KW
    g1[4] = (int)64u;                                       // tile_dim1=64, tile_dim2=0
    g1[5] = (int)(st0 & 0xFFFFFFFFull);
    g1[6] = (int)((st0 >> 32) & 0xFFFFull);                 // dim1_stride unused (2-D)
    g1[7] = 0;
    tdm_v4i gz4;
    gz4[0] = 0; gz4[1] = 0; gz4[2] = 0; gz4[3] = 0;
    tdm_v8i gz8;
#pragma unroll
    for (int i = 0; i < 8; ++i) gz8[i] = 0;
    __builtin_amdgcn_tensor_load_to_lds(g0, g1, gz4, gz4, gz8, 0);
}
#endif

// ---------------- embedding: x = wte[ids] + wpe ----------------
__global__ __launch_bounds__(256) void embed_k(const int* __restrict__ ids,
                                               const float* __restrict__ wte,
                                               const float* __restrict__ wpe,
                                               float* __restrict__ x) {
    int idx = blockIdx.x * 256 + threadIdx.x;        // < NROWS*DMODEL
    int row = idx / DMODEL;
    int col = idx - row * DMODEL;
    int s   = row & (SEQ - 1);
    x[idx] = wte[(size_t)ids[row] * DMODEL + col] + wpe[(size_t)s * DMODEL + col];
}

// ---------------- layernorm (fp32 in -> bf16 out), one wave per row ----------------
__global__ __launch_bounds__(256) void ln_k(const float* __restrict__ x,
                                            const float* __restrict__ g,
                                            const float* __restrict__ bta,
                                            unsigned short* __restrict__ out) {
    const int lane = threadIdx.x & 31;
    const int row  = blockIdx.x * 8 + (threadIdx.x >> 5);
    const float* xr = x + (size_t)row * DMODEL;
    float s = 0.0f, s2 = 0.0f;
#pragma unroll
    for (int i = lane; i < DMODEL; i += 32) { float v = xr[i]; s += v; s2 += v * v; }
#pragma unroll
    for (int off = 16; off >= 1; off >>= 1) {
        s  += __shfl_xor(s,  off, 32);
        s2 += __shfl_xor(s2, off, 32);
    }
    const float mu = s * (1.0f / DMODEL);
    const float rs = rsqrtf(s2 * (1.0f / DMODEL) - mu * mu + 1e-5f);
#pragma unroll
    for (int i = lane; i < DMODEL; i += 32)
        out[(size_t)row * DMODEL + i] = to_bf16_bits((xr[i] - mu) * rs * g[i] + bta[i]);
}

// ---------------- V transpose: qkv V-third -> vT[(b*H+h)*64+d][s] (bf16) ----------------
__global__ __launch_bounds__(256) void vtrans_k(const unsigned short* __restrict__ qkv,
                                                unsigned short* __restrict__ vT) {
    int idx = blockIdx.x * 256 + threadIdx.x;        // < NROWS*DMODEL
    int row = idx / DMODEL;                          // token index b*S + s
    int c   = idx - row * DMODEL;                    // 0..767 = h*64 + d
    int b   = row >> 10;
    int s   = row & (SEQ - 1);
    vT[((size_t)(b * DMODEL + c)) * SEQ + s] = qkv[(size_t)row * LDQKV + 2 * DMODEL + c];
}

// ---------------- generic WMMA GEMM: C[M,N] = epi(A_bf16[M,K] @ W_f32[N,K]^T + bias) ------
// block = 256 threads = 8 waves (2M x 4N), block tile 64(M) x 256(N), wave tile 32 x 64.
// A block tile staged into LDS by the Tensor Data Mover (double buffered, TENSORcnt),
// overlapping the DMA of chunk c+1 with WMMA compute on chunk c.
// EPI: 0 = store bf16, 1 = store bf16 w/ exact-erf GELU, 2 = fp32 residual +=, 3 = store fp32
template <int EPI>
__global__ __launch_bounds__(256) void gemm_k(const unsigned short* __restrict__ Ar,
                                              const float* __restrict__ W,
                                              const float* __restrict__ bias,
                                              void* __restrict__ outp,
                                              int Ndim, int Kdim) {
    const int lane = threadIdx.x & 31;
    const int wv   = threadIdx.x >> 5;
    const int lm = lane & 15, kh = lane >> 4;
    const int wm = wv & 1, wn = wv >> 1;
    const int rowBlock = blockIdx.y * 64;
    const int rowWave  = rowBlock + wm * 32;
    const int nBase    = blockIdx.x * 256 + wn * 64;

    __shared__ __align__(16) unsigned short atile[2][64 * KW];   // 2 x 16 KB

    v8f acc[2][4];
#pragma unroll
    for (int u = 0; u < 2; ++u)
#pragma unroll
        for (int t = 0; t < 4; ++t) acc[u][t] = vzero8();

    const int nch = Kdim / KW;

#if HAVE_TDM
    if (wv == 0)
        tdm_load_tile(Ar + (size_t)rowBlock * Kdim, atile[0], Kdim);
#endif

    for (int c = 0; c < nch; ++c) {
        const unsigned short* at = atile[c & 1];
#if HAVE_TDM
        if (wv == 0) __builtin_amdgcn_s_wait_tensorcnt(0);
        __syncthreads();                       // tile c visible to all waves
        if (wv == 0 && c + 1 < nch)            // prefetch next chunk via TDM
            tdm_load_tile(Ar + (size_t)rowBlock * Kdim + (c + 1) * KW,
                          atile[(c + 1) & 1], Kdim);
#else
        // cooperative staging fallback (no TDM builtin on this toolchain)
        for (int i = threadIdx.x; i < (64 * KW) / 8; i += 256) {
            int row = (i * 8) / KW;
            int col = (i * 8) - row * KW;
            ((uv4*)at)[i] =
                *(const uv4*)(Ar + (size_t)(rowBlock + row) * Kdim + c * KW + col);
        }
        __syncthreads();
#endif

        const int kg = c * KW;                 // global K offset of this chunk
#pragma unroll
        for (int kk = 0; kk < KW; kk += 32) {
            BFrag a0, a1;
            {   // A layout: lanes 0-15 K={kk..+7, kk+16..+23}, lanes 16-31 K={+8,+24}
                const uv4* p0 = (const uv4*)(at + (wm * 32 + lm) * KW + kk + kh * 8);
                a0.q[0] = p0[0];
                a0.q[1] = p0[2];
                const uv4* p1 = (const uv4*)(at + (wm * 32 + 16 + lm) * KW + kk + kh * 8);
                a1.q[0] = p1[0];
                a1.q[1] = p1[2];
            }
#pragma unroll
            for (int t = 0; t < 4; ++t) {
                BFrag b;
                const int n = nBase + t * 16 + lm;
                if (n < Ndim) {   // B column n = W row n; elements K = kg+kk + kh*8 + (j&7) + (j>>3)*16
                    const fv4* p = (const fv4*)(W + (size_t)n * Kdim + kg + kk + kh * 8);
#pragma unroll
                    for (int i = 0; i < 4; ++i) {
                        fv4 f = p[i + (i & 2)];
                        b.u[2 * i]     = pack2_bf16(f[0], f[1]);
                        b.u[2 * i + 1] = pack2_bf16(f[2], f[3]);
                    }
                } else {
#pragma unroll
                    for (int i = 0; i < 8; ++i) b.u[i] = 0u;
                }
                acc[0][t] = wmma_bf16(a0, b, acc[0][t]);
                acc[1][t] = wmma_bf16(a1, b, acc[1][t]);
            }
        }
        __syncthreads();                       // all reads of this buffer done
    }

#pragma unroll
    for (int u = 0; u < 2; ++u) {
#pragma unroll
        for (int t = 0; t < 4; ++t) {
            const int col = nBase + t * 16 + lm;
            if (col >= Ndim) continue;
            const float bv = bias ? bias[col] : 0.0f;
#pragma unroll
            for (int r = 0; r < 8; ++r) {       // C layout: row = kh*8 + r, col = lm
                const int row = rowWave + u * 16 + kh * 8 + r;
                float v = acc[u][t][r] + bv;
                const size_t o = (size_t)row * Ndim + col;
                if (EPI == 0) {
                    ((unsigned short*)outp)[o] = to_bf16_bits(v);
                } else if (EPI == 1) {
                    float gl = 0.5f * v * (1.0f + erff(v * 0.70710678118f));
                    ((unsigned short*)outp)[o] = to_bf16_bits(gl);
                } else if (EPI == 2) {
                    ((float*)outp)[o] += v;
                } else {
                    ((float*)outp)[o] = v;
                }
            }
        }
    }
}

// ---------------- flash attention: one wave per (b, h, 16-query tile) ----------------
// block = 128 (4 waves); grid = (S/16, B*H/4)
__global__ __launch_bounds__(128) void attn_k(const unsigned short* __restrict__ qkvr,
                                              const unsigned short* __restrict__ vT,
                                              unsigned short* __restrict__ aout) {
    const int lane = threadIdx.x & 31;
    const int wave = threadIdx.x >> 5;
    const int lm = lane & 15, kh = lane >> 4;
    const int qt = blockIdx.x;                 // query tile (uniform per block)
    const int bh = blockIdx.y * 4 + wave;      // 0..23 == b*NHEAD + h
    const int b  = bh / NHEAD, hh = bh % NHEAD;
    const size_t rowB = (size_t)b * SEQ;

    __shared__ __align__(16) unsigned short pst[4][16 * 32];
    unsigned short* ps = &pst[wave][0];

    const __bf16* qkv = (const __bf16*)qkvr;
    const float NEGINF = -__builtin_inff();

    // Q A-fragments for head-dim chunks d0 = 0, 32
    BFrag qa[2];
    {
        const __bf16* qrow = qkv + (rowB + qt * 16 + lm) * LDQKV + hh * HDIM;
#pragma unroll
        for (int c = 0; c < 2; ++c) {
            const uv4* p = (const uv4*)(qrow + c * 32 + kh * 16);
            qa[c].q[0] = p[0];
            qa[c].q[1] = p[1];
        }
    }

    v8f acc[4];
#pragma unroll
    for (int t = 0; t < 4; ++t) acc[t] = vzero8();
    float mrun[8], lrun[8];
#pragma unroll
    for (int r = 0; r < 8; ++r) { mrun[r] = NEGINF; lrun[r] = 0.0f; }

    const int ktmax = (qt * 16 + 15) >> 5;     // causal: key tiles of 32
    for (int kt = 0; kt <= ktmax; ++kt) {
        const int j0 = kt * 32;
        v8f s0 = vzero8(), s1 = vzero8();
#pragma unroll
        for (int c = 0; c < 2; ++c) {          // scores = Q @ K^T over HD (2 x K=32)
            const __bf16* kb = qkv + (rowB + j0) * LDQKV + DMODEL + hh * HDIM + c * 32 + kh * 16;
            BFrag b0, b1;
            const uv4* p0 = (const uv4*)(kb + (size_t)lm * LDQKV);
            b0.q[0] = p0[0]; b0.q[1] = p0[1];
            const uv4* p1 = (const uv4*)(kb + (size_t)(16 + lm) * LDQKV);
            b1.q[0] = p1[0]; b1.q[1] = p1[1];
            s0 = wmma_bf16(qa[c], b0, s0);
            s1 = wmma_bf16(qa[c], b1, s1);
        }
        // online softmax (row stats via wave-shuffle across the 16-lane half)
#pragma unroll
        for (int r = 0; r < 8; ++r) {
            const int qi = qt * 16 + kh * 8 + r;
            float v0 = (j0 + lm      <= qi) ? s0[r] * 0.125f : NEGINF;
            float v1 = (j0 + 16 + lm <= qi) ? s1[r] * 0.125f : NEGINF;
            float tm = fmaxf(v0, v1);
#pragma unroll
            for (int off = 1; off < 16; off <<= 1) tm = fmaxf(tm, __shfl_xor(tm, off, 32));
            const float mn = fmaxf(mrun[r], tm);
            const float al = __expf(mrun[r] - mn);
            mrun[r] = mn;
            const float e0 = __expf(v0 - mn);
            const float e1 = __expf(v1 - mn);
            float rs = e0 + e1;
#pragma unroll
            for (int off = 1; off < 16; off <<= 1) rs += __shfl_xor(rs, off, 32);
            lrun[r] = lrun[r] * al + rs;
#pragma unroll
            for (int t = 0; t < 4; ++t) acc[t][r] *= al;
            ps[(kh * 8 + r) * 32 + lm]      = to_bf16_bits(e0);
            ps[(kh * 8 + r) * 32 + 16 + lm] = to_bf16_bits(e1);
        }
        // P: C-layout -> A-layout via LDS (wave-private, in-order DS pipeline)
        BFrag pa;
        {
            const uv4* pp = (const uv4*)&ps[lm * 32 + kh * 16];
            pa.q[0] = pp[0];
            pa.q[1] = pp[1];
        }
        // out += P @ V : B column n = head-dim d, contiguous along keys in vT
#pragma unroll
        for (int t = 0; t < 4; ++t) {
            BFrag vb;
            const uv4* vp = (const uv4*)(vT + ((size_t)bh * HDIM + t * 16 + lm) * SEQ
                                            + j0 + kh * 16);
            vb.q[0] = vp[0];
            vb.q[1] = vp[1];
            acc[t] = wmma_bf16(pa, vb, acc[t]);
        }
    }

#pragma unroll
    for (int t = 0; t < 4; ++t) {
#pragma unroll
        for (int r = 0; r < 8; ++r) {
            const float v = acc[t][r] / lrun[r];
            const size_t row = rowB + qt * 16 + kh * 8 + r;
            aout[row * DMODEL + hh * HDIM + t * 16 + lm] = to_bf16_bits(v);
        }
    }
}

// ---------------- orchestration ----------------
extern "C" void kernel_launch(void* const* d_in, const int* in_sizes, int n_in,
                              void* d_out, int out_size, void* d_ws, size_t ws_size,
                              hipStream_t stream) {
    (void)in_sizes; (void)n_in; (void)out_size; (void)ws_size;
    const int*   ids  = (const int*)d_in[0];
    const float* wte  = (const float*)d_in[1];
    const float* wpe  = (const float*)d_in[2];
    const float* ln1g = (const float*)d_in[3];
    const float* ln1b = (const float*)d_in[4];
    const float* qkvw = (const float*)d_in[5];
    const float* qkvb = (const float*)d_in[6];
    const float* prw  = (const float*)d_in[7];
    const float* prb  = (const float*)d_in[8];
    const float* ln2g = (const float*)d_in[9];
    const float* ln2b = (const float*)d_in[10];
    const float* f1w  = (const float*)d_in[11];
    const float* f1b  = (const float*)d_in[12];
    const float* f2w  = (const float*)d_in[13];
    const float* f2b  = (const float*)d_in[14];
    const float* lnfg = (const float*)d_in[15];
    const float* lnfb = (const float*)d_in[16];
    const float* lmw  = (const float*)d_in[17];

    // workspace layout (~38 MB)
    float*          x   = (float*)d_ws;                               // [2048,768] f32 residual
    unsigned short* h   = (unsigned short*)(x + (size_t)NROWS * DMODEL);   // bf16 ln out
    unsigned short* qkv = h   + (size_t)NROWS * DMODEL;               // bf16 [2048,2304]
    unsigned short* att = qkv + (size_t)NROWS * LDQKV;                // bf16 [2048,768]
    unsigned short* ffh = att + (size_t)NROWS * DMODEL;               // bf16 [2048,3072]
    unsigned short* vT  = ffh + (size_t)NROWS * FFDIM;                // bf16 [24*64,1024]

    embed_k<<<(NROWS * DMODEL) / 256, 256, 0, stream>>>(ids, wte, wpe, x);

    for (int l = 0; l < NLAYER; ++l) {
        ln_k<<<NROWS / 8, 256, 0, stream>>>(x, ln1g + l * DMODEL, ln1b + l * DMODEL, h);
        gemm_k<0><<<dim3(LDQKV / 256, NROWS / 64), 256, 0, stream>>>(
            h, qkvw + (size_t)l * LDQKV * DMODEL, qkvb + (size_t)l * LDQKV, qkv, LDQKV, DMODEL);
        vtrans_k<<<(NROWS * DMODEL) / 256, 256, 0, stream>>>(qkv, vT);
        attn_k<<<dim3(SEQ / 16, 6), 128, 0, stream>>>(qkv, vT, att);
        gemm_k<2><<<dim3(DMODEL / 256, NROWS / 64), 256, 0, stream>>>(
            att, prw + (size_t)l * DMODEL * DMODEL, prb + (size_t)l * DMODEL, x, DMODEL, DMODEL);
        ln_k<<<NROWS / 8, 256, 0, stream>>>(x, ln2g + l * DMODEL, ln2b + l * DMODEL, h);
        gemm_k<1><<<dim3(FFDIM / 256, NROWS / 64), 256, 0, stream>>>(
            h, f1w + (size_t)l * FFDIM * DMODEL, f1b + (size_t)l * FFDIM, ffh, FFDIM, DMODEL);
        gemm_k<2><<<dim3(DMODEL / 256, NROWS / 64), 256, 0, stream>>>(
            ffh, f2w + (size_t)l * DMODEL * FFDIM, f2b + (size_t)l * DMODEL, x, DMODEL, FFDIM);
    }

    ln_k<<<NROWS / 8, 256, 0, stream>>>(x, lnfg, lnfb, h);
    gemm_k<3><<<dim3((VOCAB + 255) / 256, NROWS / 64), 256, 0, stream>>>(
        h, lmw, nullptr, d_out, VOCAB, DMODEL);
}
